// CausalSelfAttention_5205500363268
// MI455X (gfx1250) — compile-verified, matrix operand order checked by reading, not run on hardware
//
#include <hip/hip_runtime.h>
#include <hip/hip_bf16.h>

// Shapes (hardcoded from reference): B=2, T=2048, C=1024, H=16, HD=64
#define BB 2
#define TT 2048
#define CC 1024
#define HH 16
#define HD 64

typedef __attribute__((ext_vector_type(16))) _Float16 v16h;
typedef __attribute__((ext_vector_type(8)))  _Float16 v8h;
typedef __attribute__((ext_vector_type(8)))  float    v8f;

// ---------------------------------------------------------------------------
// Async global -> LDS copy (CDNA5): 16 bytes per lane, tracked by ASYNCcnt.
// LDS byte offset = low 32 bits of the generic shared address (ISA 10.2).
// ---------------------------------------------------------------------------
__device__ __forceinline__ void async_ld16(const _Float16* gsrc, _Float16* ldst) {
  unsigned int loff = (unsigned int)(unsigned long long)ldst;
  asm volatile("global_load_async_to_lds_b128 %0, %1, off"
               :: "v"(loff), "v"(gsrc) : "memory");
}
__device__ __forceinline__ void wait_async0() {
  asm volatile("s_wait_asynccnt 0x0" ::: "memory");
}

// ---------------------------------------------------------------------------
// WMMA helpers (CDNA5, wave32): D(f32 16x16) = A(f16 16x32) x B(f16 32x16) + C
// A fragment per ISA 7.12.2: lane(0..15) holds row M=lane; lanes 16..31 same
// rows, upper K-halves. Elements 0..7 = K kb..kb+7, elements 8..15 = kb+16..kb+23
// where kb = (lane>>4)*8. B fragment loads identically from an N x K (row-major)
// matrix (i.e. B^T), which matches both [out,in] weight layout and K/V tiles.
// ---------------------------------------------------------------------------
__device__ __forceinline__ v16h ldfrag(const _Float16* __restrict__ base, int ld,
                                       int row0, int k0, int lane) {
  const _Float16* p = base + (size_t)(row0 + (lane & 15)) * ld + k0 + ((lane >> 4) << 3);
  v8h lo = *(const v8h*)(p);
  v8h hi = *(const v8h*)(p + 16);
  return __builtin_shufflevector(lo, hi, 0,1,2,3,4,5,6,7,8,9,10,11,12,13,14,15);
}

__device__ __forceinline__ v8f wmma_f16(v16h a, v16h b, v8f c) {
  return __builtin_amdgcn_wmma_f32_16x16x32_f16(false, a, false, b, (short)0, c,
                                                false, false);
}

// ---------------------------------------------------------------------------
// f32 -> f16 conversion
// ---------------------------------------------------------------------------
__global__ void cvt_f32_to_f16_kernel(const float* __restrict__ src,
                                      _Float16* __restrict__ dst, int n) {
  int i = blockIdx.x * blockDim.x + threadIdx.x;
  int i4 = i * 4;
  if (i4 + 3 < n) {
    float4 v = *(const float4*)(src + i4);
    dst[i4 + 0] = (_Float16)v.x;
    dst[i4 + 1] = (_Float16)v.y;
    dst[i4 + 2] = (_Float16)v.z;
    dst[i4 + 3] = (_Float16)v.w;
  }
}

// ---------------------------------------------------------------------------
// Tiled WMMA GEMM: Out[M,N] = A[M,K] * W[N,K]^T + bias[N]
// Block tile 128(M) x 64(N), 256 threads = 8 waves in 4x2 grid, each wave 32x32.
// Double-buffered async global->LDS staging: copy of tile i+1 overlaps the
// WMMA compute of tile i; one barrier per k-step.
// MODE 0: QKV -> scatter into head-major f16 Q/K/V [B,H,T,HD]
// MODE 1: f32 output [M,N] (final projection)
// ---------------------------------------------------------------------------
template <int MODE>
__global__ __launch_bounds__(256)
void wmma_gemm_kernel(const _Float16* __restrict__ A,
                      const _Float16* __restrict__ W,
                      const float* __restrict__ bias,
                      _Float16* __restrict__ Qh, _Float16* __restrict__ Kh,
                      _Float16* __restrict__ Vh, float* __restrict__ Out,
                      int K) {
  const int tid  = threadIdx.x;
  const int lane = tid & 31;
  const int w    = tid >> 5;      // 0..7
  const int wm   = w >> 1;        // 0..3
  const int wn   = w & 1;         // 0..1
  const int m0   = blockIdx.x * 128;
  const int n0   = blockIdx.y * 64;

  __shared__ _Float16 As[2][128 * 40];
  __shared__ _Float16 Bs[2][64 * 40];

  v8f acc[2][2] = {};

  const int arow = tid >> 1;          // 0..127
  const int acol = (tid & 1) * 16;    // 0,16
  const int brow = tid >> 2;          // 0..63
  const int bcol = (tid & 3) * 8;     // 0,8,16,24

  auto issue_tile = [&](int k0, int buf) {
    const _Float16* ag = A + (size_t)(m0 + arow) * K + k0 + acol;
    async_ld16(ag,     &As[buf][arow * 40 + acol]);
    async_ld16(ag + 8, &As[buf][arow * 40 + acol + 8]);
    const _Float16* bg = W + (size_t)(n0 + brow) * K + k0 + bcol;
    async_ld16(bg,     &Bs[buf][brow * 40 + bcol]);
    if (k0 + 32 < K) {  // pull the tile after next toward L2 (global_prefetch_b8)
      __builtin_prefetch(ag + 32, 0, 1);
      __builtin_prefetch(bg + 32, 0, 1);
    }
  };

  const int nk = K >> 5;   // k-steps of 32
  issue_tile(0, 0);
  for (int i = 0; i < nk; ++i) {
    wait_async0();          // tile i landed in LDS (per-wave), then make visible
    __syncthreads();
    if (i + 1 < nk) issue_tile((i + 1) << 5, (i + 1) & 1);  // overlap with compute

    const _Float16* as = As[i & 1];
    const _Float16* bs = Bs[i & 1];
    v16h af0 = ldfrag(as, 40, wm * 32,      0, lane);
    v16h af1 = ldfrag(as, 40, wm * 32 + 16, 0, lane);
    v16h bf0 = ldfrag(bs, 40, wn * 32,      0, lane);
    v16h bf1 = ldfrag(bs, 40, wn * 32 + 16, 0, lane);
    acc[0][0] = wmma_f16(af0, bf0, acc[0][0]);
    acc[0][1] = wmma_f16(af0, bf1, acc[0][1]);
    acc[1][0] = wmma_f16(af1, bf0, acc[1][0]);
    acc[1][1] = wmma_f16(af1, bf1, acc[1][1]);
  }

  // D layout: lane holds column n = lane&15; element e is row (lane>>4)*8 + e
  const int rb = (lane >> 4) << 3;
  const int nn = lane & 15;
#pragma unroll
  for (int i = 0; i < 2; ++i) {
#pragma unroll
    for (int j = 0; j < 2; ++j) {
      const int mt = wm * 32 + i * 16;
      const int nt = wn * 32 + j * 16;
#pragma unroll
      for (int e = 0; e < 8; ++e) {
        const int row = m0 + mt + rb + e;      // token index b*T + t
        const int col = n0 + nt + nn;          // output feature
        float v = acc[i][j][e] + bias[col];
        if (MODE == 0) {
          const int sel = col >> 10;           // 0=q 1=k 2=v
          const int c   = col & (CC - 1);
          const int h   = c >> 6;
          const int d   = c & (HD - 1);
          const int b   = row >> 11;
          const int t   = row & (TT - 1);
          const size_t di = ((size_t)(b * HH + h) * TT + t) * HD + d;
          const _Float16 hv = (_Float16)v;
          if (sel == 0)      Qh[di] = hv;
          else if (sel == 1) Kh[di] = hv;
          else               Vh[di] = hv;
        } else {
          Out[(size_t)row * CC + col] = v;
        }
      }
    }
  }
}

// ---------------------------------------------------------------------------
// In-place RoPE on head-major f16 Q and K. One thread per (bh,t,pair).
// ---------------------------------------------------------------------------
__global__ void rope_kernel(_Float16* __restrict__ Qh, _Float16* __restrict__ Kh,
                            const float* __restrict__ cosT,
                            const float* __restrict__ sinT) {
  const int idx = blockIdx.x * blockDim.x + threadIdx.x;
  const int total = BB * HH * TT * (HD / 2);
  if (idx >= total) return;
  const int p = idx & 31;
  const int t = (idx >> 5) & (TT - 1);
  const float c = cosT[t * 32 + p];
  const float s = sinT[t * 32 + p];
  const size_t off = (size_t)idx * 2;
  float q0 = (float)Qh[off], q1 = (float)Qh[off + 1];
  Qh[off]     = (_Float16)(q0 * c - q1 * s);
  Qh[off + 1] = (_Float16)(q0 * s + q1 * c);
  float k0 = (float)Kh[off], k1 = (float)Kh[off + 1];
  Kh[off]     = (_Float16)(k0 * c - k1 * s);
  Kh[off + 1] = (_Float16)(k0 * s + k1 * c);
}

// ---------------------------------------------------------------------------
// Flash attention: grid = (T/64 q-tiles, B*H). 128 threads = 4 waves, each wave
// owns 16 q rows. Streams causal key tiles of 64 with online softmax.
// K tile double-buffered via async global->LDS (copy of kt+1 overlaps the whole
// scores/softmax/PV phase of kt); V staged synchronously (transposed in LDS).
// Scores:  S = Q(16xHD) * Ktile^T     (Ktile rows are keys -> N x K row-major)
// PV:      O += P(16x64) * Vtile      (V transposed in LDS -> N x K row-major)
// ---------------------------------------------------------------------------
__global__ __launch_bounds__(128)
void flash_attn_kernel(const _Float16* __restrict__ Qh,
                       const _Float16* __restrict__ Kh,
                       const _Float16* __restrict__ Vh,
                       _Float16* __restrict__ Ao) {
  const int tid  = threadIdx.x;
  const int lane = tid & 31;
  const int w    = tid >> 5;          // wave 0..3
  const int qt   = blockIdx.x;        // q tile (64 rows)
  const int bh   = blockIdx.y;        // fused batch*head
  const int bb   = bh >> 4;
  const int hh   = bh & (HH - 1);

  __shared__ _Float16 Kt[2][64 * 72];     // [key][hd], double buffered
  __shared__ _Float16 Vt[64 * 72];        // [hd][key] (transposed)
  __shared__ float    Ss[4][16 * 65];     // per-wave scores
  __shared__ _Float16 Ps[4][16 * 72];     // per-wave probs (f16)
  __shared__ float    sc[4][16];          // per-row scale broadcast

  const size_t headoff = (size_t)bh * TT * HD;
  const _Float16* Qp = Qh + headoff;
  const _Float16* Kp = Kh + headoff;
  const _Float16* Vp = Vh + headoff;

  const int qrow0 = qt * 64 + w * 16;     // this wave's q-row base
  const v16h qf0 = ldfrag(Qp + (size_t)qrow0 * HD, HD, 0, 0,  lane);
  const v16h qf1 = ldfrag(Qp + (size_t)qrow0 * HD, HD, 0, 32, lane);

  v8f Oa[4] = {};                         // 4 n-tiles covering HD=64
  float mrow = -1e30f, lrow = 0.0f;       // online softmax state (lanes 0..15)

  const int rb = (lane >> 4) << 3;
  const int nn = lane & 15;
  const int nkt = qt + 1;                 // causal bound

  const int r  = tid >> 1;                // 0..63 (staging row)
  const int cb = (tid & 1) * 32;          // 0,32

  auto issue_k = [&](int kt, int buf) {
    const _Float16* kg = Kp + (size_t)(kt * 64 + r) * HD + cb;
    async_ld16(kg,      &Kt[buf][r * 72 + cb]);
    async_ld16(kg + 8,  &Kt[buf][r * 72 + cb + 8]);
    async_ld16(kg + 16, &Kt[buf][r * 72 + cb + 16]);
    async_ld16(kg + 24, &Kt[buf][r * 72 + cb + 24]);
  };

  issue_k(0, 0);
  for (int kt = 0; kt < nkt; ++kt) {
    // --- stage V tile transposed via VGPRs ---------------------------------
    {
      const _Float16* vg = Vp + (size_t)(kt * 64 + r) * HD + cb;
      v8h v0 = *(const v8h*)(vg);
      v8h v1 = *(const v8h*)(vg + 8);
      v8h v2 = *(const v8h*)(vg + 16);
      v8h v3 = *(const v8h*)(vg + 24);
#pragma unroll
      for (int j = 0; j < 8; ++j) {
        Vt[(cb + j)      * 72 + r] = v0[j];
        Vt[(cb + 8 + j)  * 72 + r] = v1[j];
        Vt[(cb + 16 + j) * 72 + r] = v2[j];
        Vt[(cb + 24 + j) * 72 + r] = v3[j];
      }
    }
    wait_async0();            // K tile kt landed (per-wave)
    __syncthreads();          // Kt[kt&1] + Vt visible block-wide
    if (kt + 1 < nkt) issue_k(kt + 1, (kt + 1) & 1);  // overlap with compute

    const _Float16* ktile = Kt[kt & 1];

    // --- scores: 4 x (16x16) tiles, scale + causal mask -> LDS --------------
#pragma unroll
    for (int nt = 0; nt < 4; ++nt) {
      v8f s = {};
      v16h b0 = ldfrag(ktile, 72, nt * 16, 0,  lane);
      v16h b1 = ldfrag(ktile, 72, nt * 16, 32, lane);
      s = wmma_f16(qf0, b0, s);
      s = wmma_f16(qf1, b1, s);
      const int kcol = kt * 64 + nt * 16 + nn;
#pragma unroll
      for (int e = 0; e < 8; ++e) {
        const int qrow = qrow0 + rb + e;
        float v = s[e] * 0.125f;          // HD^-0.5
        if (kcol > qrow) v = -1e30f;
        Ss[w][(rb + e) * 65 + nt * 16 + nn] = v;
      }
    }
    __syncthreads();

    // --- online softmax: lane r (<16) owns q-row r of this wave -------------
    if (lane < 16) {
      const float* srow = &Ss[w][lane * 65];
      float tmax = -1e30f;
#pragma unroll 8
      for (int j = 0; j < 64; ++j) tmax = fmaxf(tmax, srow[j]);
      const float mnew  = fmaxf(mrow, tmax);
      const float alpha = __expf(mrow - mnew);
      float sum = 0.0f;
      _Float16* prow = &Ps[w][lane * 72];
#pragma unroll 8
      for (int j = 0; j < 64; ++j) {
        const float p = __expf(srow[j] - mnew);
        sum += p;
        prow[j] = (_Float16)p;
      }
      lrow = alpha * lrow + sum;
      mrow = mnew;
      sc[w][lane] = alpha;
    }
    __syncthreads();

    // --- rescale O, then O += P * V -----------------------------------------
#pragma unroll
    for (int nt = 0; nt < 4; ++nt) {
#pragma unroll
      for (int e = 0; e < 8; ++e) Oa[nt][e] *= sc[w][rb + e];
    }
    const v16h pa0 = ldfrag(Ps[w], 72, 0, 0,  lane);
    const v16h pa1 = ldfrag(Ps[w], 72, 0, 32, lane);
#pragma unroll
    for (int nt = 0; nt < 4; ++nt) {
      v16h vb0 = ldfrag(Vt, 72, nt * 16, 0,  lane);
      v16h vb1 = ldfrag(Vt, 72, nt * 16, 32, lane);
      Oa[nt] = wmma_f16(pa0, vb0, Oa[nt]);
      Oa[nt] = wmma_f16(pa1, vb1, Oa[nt]);
    }
    __syncthreads();    // protect Vt (and P/S) before next tile's overwrite
  }

  // --- epilogue: O / l, write token-major f16 [B,T,C] -----------------------
  if (lane < 16) sc[w][lane] = 1.0f / lrow;
  __syncthreads();
#pragma unroll
  for (int nt = 0; nt < 4; ++nt) {
#pragma unroll
    for (int e = 0; e < 8; ++e) {
      const int qrow = qrow0 + rb + e;
      const float v = Oa[nt][e] * sc[w][rb + e];
      Ao[((size_t)(bb * TT + qrow)) * CC + hh * HD + nt * 16 + nn] = (_Float16)v;
    }
  }
}

// ---------------------------------------------------------------------------
// Launcher
// ---------------------------------------------------------------------------
extern "C" void kernel_launch(void* const* d_in, const int* in_sizes, int n_in,
                              void* d_out, int out_size, void* d_ws, size_t ws_size,
                              hipStream_t stream) {
  const float* x      = (const float*)d_in[0];   // [B,T,C]
  const float* fcos   = (const float*)d_in[1];   // [T,HD/2]
  const float* fsin   = (const float*)d_in[2];   // [T,HD/2]
  const float* qkv_w  = (const float*)d_in[3];   // [3C,C]
  const float* qkv_b  = (const float*)d_in[4];   // [3C]
  const float* o_w    = (const float*)d_in[5];   // [C,C]
  const float* o_b    = (const float*)d_in[6];   // [C]
  float* out = (float*)d_out;                    // [B,T,C]

  char* ws = (char*)d_ws;
  const size_t nX    = (size_t)BB * TT * CC;       // 4194304
  const size_t nWq   = (size_t)3 * CC * CC;        // 3145728
  const size_t nWo   = (size_t)CC * CC;            // 1048576
  const size_t nHead = (size_t)BB * HH * TT * HD;  // 4194304

  _Float16* xh    = (_Float16*)(ws);
  _Float16* wqkvh = (_Float16*)(ws + 2 * nX);
  _Float16* woh   = (_Float16*)(ws + 2 * (nX + nWq));
  _Float16* Qh    = (_Float16*)(ws + 2 * (nX + nWq + nWo));
  _Float16* Kh    = (_Float16*)(ws + 2 * (nX + nWq + nWo + nHead));
  _Float16* Vh    = (_Float16*)(ws + 2 * (nX + nWq + nWo + 2 * nHead));
  _Float16* aoh   = (_Float16*)(ws + 2 * (nX + nWq + nWo + 3 * nHead));

  // 1) f32 -> f16 staging
  cvt_f32_to_f16_kernel<<<(int)(nX / 4 + 255) / 256, 256, 0, stream>>>(x, xh, (int)nX);
  cvt_f32_to_f16_kernel<<<(int)(nWq / 4 + 255) / 256, 256, 0, stream>>>(qkv_w, wqkvh, (int)nWq);
  cvt_f32_to_f16_kernel<<<(int)(nWo / 4 + 255) / 256, 256, 0, stream>>>(o_w, woh, (int)nWo);

  // 2) QKV projection + bias, scattered to head-major Q/K/V
  wmma_gemm_kernel<0><<<dim3(32, 48), 256, 0, stream>>>(
      xh, wqkvh, qkv_b, Qh, Kh, Vh, nullptr, CC);

  // 3) RoPE in place on Q and K
  rope_kernel<<<(int)(nHead / 2 + 255) / 256, 256, 0, stream>>>(Qh, Kh, fcos, fsin);

  // 4) fused causal flash attention -> f16 [B,T,C]
  flash_attn_kernel<<<dim3(TT / 64, BB * HH), 128, 0, stream>>>(Qh, Kh, Vh, aoh);

  // 5) output projection + bias -> f32 out
  wmma_gemm_kernel<1><<<dim3(32, 16), 256, 0, stream>>>(
      aoh, woh, o_b, nullptr, nullptr, nullptr, out, CC);
}
